// AttentionModel_30657476559105
// MI455X (gfx1250) — matrix operand order repaired and checked by reading, not compiled, not run on hardware
//
#include <hip/hip_runtime.h>

typedef __attribute__((ext_vector_type(16))) _Float16 v16h;
typedef __attribute__((ext_vector_type(8)))  _Float16 v8h;
typedef __attribute__((ext_vector_type(8)))  float    v8f;

// B=4, Q=65536, HID=64, IN=8, 3 tokens. HID collapses algebraically:
//   M[a][b] = sum_h wq[h][a]*wk[h][b], c1 = Wq^T bk, c2 = bq^T Wk, c0 = bq.bk
//   P[a][o] = sum_h wv[h][a]*wo[o][h], d  = Wo bv
//   scores_ij = (x_i M x_j^T + x_i.c1 + c2.x_j + c0)/8 ; softmax rows sum to 1
//   out_io = sum_j w_ij (x_j P[:,o]) + d[o] + bo[o] ; final_o = sum_i w[i][o]*out[i][o]
// These K=64 weight contractions are computed per-block via v_wmma_f32_16x16x32_f16.

__global__ __launch_bounds__(256) void attn_fused_kernel(
    const float* __restrict__ preds, const float* __restrict__ dist,
    const float* __restrict__ wq, const float* __restrict__ bq,
    const float* __restrict__ wk, const float* __restrict__ bk,
    const float* __restrict__ wv, const float* __restrict__ bv,
    const float* __restrict__ wo, const float* __restrict__ bo,
    float* __restrict__ out, int nQ)
{
    // f16 operand tables, lane-major so each WMMA fragment = 2x ds_load_b128
    __shared__ _Float16 sAq[16][64];   // [m][k] : Wq^T, row8 = bq, rows 9-15 = 0
    __shared__ _Float16 sBk[16][64];   // [n][k] : Wk,   row8 = bk, rows 9-15 = 0
    __shared__ _Float16 sAv[16][64];   // [m][k] : Wv^T, row8 = bv
    __shared__ _Float16 sBw[16][64];   // [n][k] : Wo^T (rows 0-2), rest 0
    __shared__ float sM[9][9];         // M; row 8 = c2, col 8 = c1, [8][8] = c0
    __shared__ float sP[9][3];         // P; row 8 = d
    __shared__ float sC[3];            // bo

    const int tid = threadIdx.x;

    // ---- Phase 1: cooperative staging (branch-light, parallel) ----
    #pragma unroll
    for (int it = 0; it < 4; ++it) {
        const int idx = tid + it * 256;
        const int row = idx >> 6;
        const int k   = idx & 63;
        float vq = 0.0f, vk = 0.0f, vv = 0.0f, vo = 0.0f;
        if (row < 8)       { vq = wq[k*8+row]; vk = wk[k*8+row]; vv = wv[k*8+row]; }
        else if (row == 8) { vq = bq[k];       vk = bk[k];       vv = bv[k]; }
        if (row < 3)       { vo = wo[row*64 + k]; }
        sAq[row][k] = (_Float16)vq;
        sBk[row][k] = (_Float16)vk;
        sAv[row][k] = (_Float16)vv;
        sBw[row][k] = (_Float16)vo;
    }
    if (tid < 3) sC[tid] = bo[tid];
    __syncthreads();

    // ---- Phase 2: wave 0 contracts the weights with WMMA (K=64 in 2 steps) ----
    if (tid < 32) {
        const int col = tid & 15;   // A: row m / B,C: column n
        const int hi  = tid >> 4;
        const _Float16* aq = &sAq[col][0];
        const _Float16* bkp = &sBk[col][0];
        const _Float16* av = &sAv[col][0];
        const _Float16* bw = &sBw[col][0];
        v8f acc1 = {}, acc2 = {};
        #pragma unroll
        for (int kc = 0; kc < 64; kc += 32) {
            // A 16x32 f16 layout: slots 0-7 -> K = kc+hi*8..+7 ; slots 8-15 -> +16
            const int ka = kc + hi * 8;
            v8h a1l = *reinterpret_cast<const v8h*>(aq + ka);
            v8h a1h = *reinterpret_cast<const v8h*>(aq + ka + 16);
            v8h a2l = *reinterpret_cast<const v8h*>(av + ka);
            v8h a2h = *reinterpret_cast<const v8h*>(av + ka + 16);
            // B 32x16 f16 layout: slots 0-7 -> K = kc+hi*8..+7 ; slots 8-15 -> +16
            const int kb = kc + hi * 8;
            v8h b1l = *reinterpret_cast<const v8h*>(bkp + kb);
            v8h b1h = *reinterpret_cast<const v8h*>(bkp + kb + 16);
            v8h b2l = *reinterpret_cast<const v8h*>(bw + kb);
            v8h b2h = *reinterpret_cast<const v8h*>(bw + kb + 16);
            v16h a1 = __builtin_shufflevector(a1l, a1h, 0,1,2,3,4,5,6,7,8,9,10,11,12,13,14,15);
            v16h b1 = __builtin_shufflevector(b1l, b1h, 0,1,2,3,4,5,6,7,8,9,10,11,12,13,14,15);
            v16h a2 = __builtin_shufflevector(a2l, a2h, 0,1,2,3,4,5,6,7,8,9,10,11,12,13,14,15);
            v16h b2 = __builtin_shufflevector(b2l, b2h, 0,1,2,3,4,5,6,7,8,9,10,11,12,13,14,15);
            acc1 = __builtin_amdgcn_wmma_f32_16x16x32_f16(false, a1, false, b1,
                                                          (short)0, acc1, false, false);
            acc2 = __builtin_amdgcn_wmma_f32_16x16x32_f16(false, a2, false, b2,
                                                          (short)0, acc2, false, false);
        }
        // C/D layout: VGPR r -> row r (lanes 0-15) or 8+r (lanes 16-31); lane -> col
        if (hi == 0 && col < 9) {
            #pragma unroll
            for (int r = 0; r < 8; ++r) sM[r][col] = acc1[r];
        }
        if (hi == 1 && col < 9) sM[8][col] = acc1[0];
        if (hi == 0 && col < 3) {
            #pragma unroll
            for (int r = 0; r < 8; ++r) sP[r][col] = acc2[r];
        }
        if (hi == 1 && col < 3) sP[8][col] = acc2[0];
    }
    __syncthreads();

    // ---- Phase 3: stream-bound main loop, 4 queries per thread ----
    const int stride = gridDim.x * blockDim.x;
    for (int q = blockIdx.x * blockDim.x + tid; q < nQ; q += stride) {
        // speculative prefetch one stride ahead (safe if OOB)
        __builtin_prefetch(&preds[(size_t)(q + stride) * 12], 0, 0);
        __builtin_prefetch(&dist[(size_t)(q + stride) * 4], 0, 0);

        const float4 p0 = reinterpret_cast<const float4*>(preds)[q * 3 + 0];
        const float4 p1 = reinterpret_cast<const float4*>(preds)[q * 3 + 1];
        const float4 p2 = reinterpret_cast<const float4*>(preds)[q * 3 + 2];
        const float4 dd = reinterpret_cast<const float4*>(dist)[q];

        float x[3][8] = {
            {p0.x, p0.y, p0.z, p0.w, dd.x, dd.y, dd.z, dd.w},
            {p1.x, p1.y, p1.z, p1.w, dd.x, dd.y, dd.z, dd.w},
            {p2.x, p2.y, p2.z, p2.w, dd.x, dd.y, dd.z, dd.w}};

        const float c0 = sM[8][8];
        float f[3];
        #pragma unroll
        for (int j = 0; j < 3; ++j) {
            float acc = 0.0f;
            #pragma unroll
            for (int b = 0; b < 8; ++b) acc += sM[8][b] * x[j][b];
            f[j] = acc;
        }

        float w[3][3];
        #pragma unroll
        for (int i = 0; i < 3; ++i) {
            float t[8];
            #pragma unroll
            for (int b = 0; b < 8; ++b) {
                float acc = 0.0f;
                #pragma unroll
                for (int a = 0; a < 8; ++a) acc += x[i][a] * sM[a][b];
                t[b] = acc;
            }
            float ei = 0.0f;
            #pragma unroll
            for (int a = 0; a < 8; ++a) ei += x[i][a] * sM[a][8];
            float s[3];
            #pragma unroll
            for (int j = 0; j < 3; ++j) {
                float acc = 0.0f;
                #pragma unroll
                for (int b = 0; b < 8; ++b) acc += t[b] * x[j][b];
                s[j] = (acc + ei + f[j] + c0) * 0.125f;   // 1/sqrt(HID)
            }
            const float mx = fmaxf(s[0], fmaxf(s[1], s[2]));
            const float e0 = __expf(s[0] - mx);
            const float e1 = __expf(s[1] - mx);
            const float e2 = __expf(s[2] - mx);
            const float inv = 1.0f / (e0 + e1 + e2);
            w[i][0] = e0 * inv; w[i][1] = e1 * inv; w[i][2] = e2 * inv;
        }

        float y[3][3];
        #pragma unroll
        for (int j = 0; j < 3; ++j)
            #pragma unroll
            for (int o = 0; o < 3; ++o) {
                float acc = sP[8][o] + sC[o];
                #pragma unroll
                for (int a = 0; a < 8; ++a) acc += x[j][a] * sP[a][o];
                y[j][o] = acc;
            }

        float fin[3] = {0.0f, 0.0f, 0.0f};
        #pragma unroll
        for (int i = 0; i < 3; ++i)
            #pragma unroll
            for (int o = 0; o < 3; ++o) {
                const float oio = w[i][0] * y[0][o] + w[i][1] * y[1][o] + w[i][2] * y[2][o];
                fin[o] += w[i][o] * oio;
            }

        out[q * 3 + 0] = fin[0];
        out[q * 3 + 1] = fin[1];
        out[q * 3 + 2] = fin[2];
    }
}

extern "C" void kernel_launch(void* const* d_in, const int* in_sizes, int n_in,
                              void* d_out, int out_size, void* d_ws, size_t ws_size,
                              hipStream_t stream) {
    (void)n_in; (void)d_ws; (void)ws_size; (void)out_size;
    const float* preds = (const float*)d_in[0];
    const float* dist  = (const float*)d_in[1];
    const float* wq    = (const float*)d_in[2];
    const float* bq    = (const float*)d_in[3];
    const float* wk    = (const float*)d_in[4];
    const float* bk    = (const float*)d_in[5];
    const float* wv    = (const float*)d_in[6];
    const float* bv    = (const float*)d_in[7];
    const float* wo    = (const float*)d_in[8];
    const float* bo    = (const float*)d_in[9];

    const int nQ = in_sizes[0] / 12;           // B*Q (3*4 floats per query)
    const int threads = 256;                   // 8 wave32s
    const int qPerThread = 4;                  // amortize preamble, add MLP
    int blocks = (nQ + threads * qPerThread - 1) / (threads * qPerThread);
    if (blocks < 1) blocks = 1;
    attn_fused_kernel<<<blocks, threads, 0, stream>>>(
        preds, dist, wq, bq, wk, bk, wv, bv, wo, bo, (float*)d_out, nQ);
}
